// FSMNLayer_18382460027457
// MI455X (gfx1250) — matrix operand order repaired and checked by reading, not compiled, not run on hardware
//
#include <hip/hip_runtime.h>

// ---------------------------------------------------------------------------
// FSMN layer for MI455X (gfx1250, wave32, WMMA):
//   out = x @ W_lin^T + sliding_mean(x) @ W_mem^T + (b_lin + b_mem)
// Fused as one [B*T, 512] x [512, 256] GEMM in bf16 with f32 accumulation.
// B=16, T=8192, H=256 fixed by the reference.
// ---------------------------------------------------------------------------

typedef __attribute__((ext_vector_type(16))) __bf16 v16bf;
typedef __attribute__((ext_vector_type(8)))  __bf16 bf16x8;
typedef __attribute__((ext_vector_type(8)))  float  v8f;

static constexpr int H      = 256;
static constexpr int Bsz    = 16;
static constexpr int T      = 8192;
static constexpr int STRIDE = 8;     // window [max(0,t-8), t] -> up to 9 taps
static constexpr int KCAT   = 2 * H; // 512: [x | means]
static constexpr int TCHUNK = 128;

__device__ __forceinline__ unsigned short f32_to_bf16(float f) {
    unsigned int u = __float_as_uint(f);
    u += 0x7fffu + ((u >> 16) & 1u);   // round-to-nearest-even
    return (unsigned short)(u >> 16);
}

// ---------------------------------------------------------------------------
// Stage 0: pack W_cat = [W_lin | W_mem] (bf16, row-major [o][k], k in 0..511)
//          and bias_cat = b_lin + b_mem (f32).
// ---------------------------------------------------------------------------
__global__ __launch_bounds__(256) void fsmn_prep_w(
    const float* __restrict__ Wl, const float* __restrict__ bl,
    const float* __restrict__ Wm, const float* __restrict__ bm,
    unsigned short* __restrict__ wcat, float* __restrict__ bias) {
    int idx = blockIdx.x * 256 + threadIdx.x;   // 0 .. 256*512-1
    int o = idx >> 9;
    int k = idx & (KCAT - 1);
    float v = (k < H) ? Wl[o * H + k] : Wm[o * H + (k - H)];
    wcat[idx] = f32_to_bf16(v);
    if (k == 0) bias[o] = bl[o] + bm[o];
}

// ---------------------------------------------------------------------------
// Stage 1: running-sum sliding mean + bf16 pack.
// Thread h (0..255) walks TCHUNK timesteps of one batch row: each x element is
// read at most twice (add + retire), writes xcat[row][0:256]=x, [256:512]=mean.
// Loads/stores are fully coalesced (thread h <-> consecutive addresses).
// ---------------------------------------------------------------------------
__global__ __launch_bounds__(256) void fsmn_means_pack(
    const float* __restrict__ x, unsigned short* __restrict__ xcat) {
    const int h      = threadIdx.x;
    const int nchunk = T / TCHUNK;
    const int b      = blockIdx.x / nchunk;
    const int t0     = (blockIdx.x % nchunk) * TCHUNK;

    const float*    xb = x    + (size_t)b * T * H + h;
    unsigned short* yb = xcat + (size_t)b * T * KCAT + h;

    float s = 0.0f;                       // warm-up: window tail [t0-8, t0-1]
    for (int t = t0 - STRIDE; t < t0; ++t)
        if (t >= 0) s += xb[(size_t)t * H];

    for (int t = t0; t < t0 + TCHUNK; ++t) {
        float xv = xb[(size_t)t * H];
        s += xv;                          // s now covers [max(0,t-8), t]
        float cnt  = (float)((t < STRIDE) ? (t + 1) : (STRIDE + 1));
        float mean = s / cnt;
        yb[(size_t)t * KCAT]     = f32_to_bf16(xv);
        yb[(size_t)t * KCAT + H] = f32_to_bf16(mean);
        int told = t - STRIDE;            // retire oldest tap
        if (told >= 0) s -= xb[(size_t)told * H];
    }
}

// ---------------------------------------------------------------------------
// Stage 2: WMMA GEMM. Block = 8 waves, M-tile = 16 rows, N = full 256.
// Wave w computes N-tiles {2w, 2w+1}; A fragment shared across both WMMAs.
// K loop: 512 / 32 = 16 iterations, fully unrolled -> 32 v_wmma per wave.
// ---------------------------------------------------------------------------
__global__ __launch_bounds__(256) void fsmn_wmma_gemm(
    const unsigned short* __restrict__ xcat,
    const unsigned short* __restrict__ wcat,
    const float* __restrict__ bias,
    float* __restrict__ out) {
    const int lane = threadIdx.x & 31;
    const int wave = threadIdx.x >> 5;
    const int m    = lane & 15;          // A row within tile
    const int g    = lane >> 4;          // lane half selects K sub-ranges

    const size_t row0 = (size_t)blockIdx.x * 16;

    // A: row-major [row][512] bf16. Per ISA 16-bit A layout:
    //   elements 0..7  <-> K = g*8 + i,      elements 8..15 <-> K = 16 + g*8 + i
    const __bf16* Aptr = (const __bf16*)xcat + (row0 + m) * KCAT;

    // B: B[k][n] = W_cat[n][k]; lane holds N = lane&15, K = g*8 + 0..7 and 16 + g*8 + 0..7
    const int n_loc = lane & 15;
    const int n0    = wave * 32 + n_loc;
    const int n1    = n0 + 16;
    const __bf16* B0ptr = (const __bf16*)wcat + (size_t)n0 * KCAT + g * 8;
    const __bf16* B1ptr = (const __bf16*)wcat + (size_t)n1 * KCAT + g * 8;

    v8f acc0 = {};
    v8f acc1 = {};

#pragma unroll
    for (int k0 = 0; k0 < KCAT; k0 += 32) {
        bf16x8 a_lo = *(const bf16x8*)(Aptr + k0 + g * 8);
        bf16x8 a_hi = *(const bf16x8*)(Aptr + k0 + 16 + g * 8);
        v16bf a;
#pragma unroll
        for (int i = 0; i < 8; ++i) { a[i] = a_lo[i]; a[i + 8] = a_hi[i]; }

        bf16x8 b0_lo = *(const bf16x8*)(B0ptr + k0);
        bf16x8 b0_hi = *(const bf16x8*)(B0ptr + k0 + 16);
        bf16x8 b1_lo = *(const bf16x8*)(B1ptr + k0);
        bf16x8 b1_hi = *(const bf16x8*)(B1ptr + k0 + 16);
        v16bf b0, b1;
#pragma unroll
        for (int i = 0; i < 8; ++i) {
            b0[i] = b0_lo[i]; b0[i + 8] = b0_hi[i];
            b1[i] = b1_lo[i]; b1[i + 8] = b1_hi[i];
        }

        acc0 = __builtin_amdgcn_wmma_f32_16x16x32_bf16(
            false, a, false, b0, (short)0, acc0, false, false);
        acc1 = __builtin_amdgcn_wmma_f32_16x16x32_bf16(
            false, a, false, b1, (short)0, acc1, false, false);
    }

    const float bz0 = bias[n0];
    const float bz1 = bias[n1];
    float* orow = out + row0 * H;
    // C/D layout: element r <-> M = g*8 + r, N = lane&15
#pragma unroll
    for (int r = 0; r < 8; ++r) {
        const int mr = g * 8 + r;
        orow[(size_t)mr * H + n0] = acc0[r] + bz0;
        orow[(size_t)mr * H + n1] = acc1[r] + bz1;
    }
}

// ---------------------------------------------------------------------------
extern "C" void kernel_launch(void* const* d_in, const int* in_sizes, int n_in,
                              void* d_out, int out_size, void* d_ws, size_t ws_size,
                              hipStream_t stream) {
    (void)in_sizes; (void)n_in; (void)out_size; (void)ws_size;
    const float* x   = (const float*)d_in[0];   // [16, 8192, 256]
    const float* Wl  = (const float*)d_in[1];   // [256, 256]
    const float* bl  = (const float*)d_in[2];   // [256]
    const float* Wm  = (const float*)d_in[3];   // [256, 256]
    const float* bm  = (const float*)d_in[4];   // [256]
    float*       out = (float*)d_out;           // [16, 8192, 256]

    // Workspace layout (needs ~134.5 MB):
    //   xcat : B*T*512 bf16  = 134,217,728 B
    //   wcat : 256*512 bf16  =     262,144 B
    //   bias : 256 f32       =       1,024 B
    unsigned short* xcat = (unsigned short*)d_ws;
    unsigned short* wcat = xcat + (size_t)Bsz * T * KCAT;
    float*          bias = (float*)(wcat + (size_t)H * KCAT);

    fsmn_prep_w<<<(H * KCAT) / 256, 256, 0, stream>>>(Wl, bl, Wm, bm, wcat, bias);
    fsmn_means_pack<<<Bsz * (T / TCHUNK), 256, 0, stream>>>(x, xcat);
    fsmn_wmma_gemm<<<(Bsz * T) / 16, 256, 0, stream>>>(xcat, wcat, bias, out);
}
